// PixToShapeCycleLoss_82231443849658
// MI455X (gfx1250) — matrix operand order repaired and checked by reading, not compiled, not run on hardware
//
#include <hip/hip_runtime.h>
#include <hip/hip_bf16.h>
#include <math.h>

// Problem constants (from reference)
#define NN 16      // instances
#define DD 16      // embed dim
#define MM 2       // meshes
#define VV 7829    // vertices
#define KK 512     // sampled pixels
#define SS 112     // heatmap size
#define VP 7872    // VV padded to multiple of 64 (123 * 64)
#define INV_T 20.0f

// LDS panel geometry for the WMMA kernel (halves)
#define ROWH 72                    // 64 data halves + 8 pad (144 B rows, 16B aligned)
#define PANEL_H (64 * ROWH)        // halves per panel (A or B)
#define PANEL_B (PANEL_H * 2)      // bytes per panel = 9216
#define BUF_B (2 * PANEL_B)        // A+B panels per buffer = 18432
#define LDS_TOTAL (2 * BUF_B)      // double buffered = 36864 B

typedef _Float16 v16h __attribute__((ext_vector_type(16)));
typedef _Float16 hv8  __attribute__((ext_vector_type(8)));
typedef float    v8f  __attribute__((ext_vector_type(8)));

// CDNA5 async global->LDS copy (ASYNCcnt path), 16 B per lane.
__device__ __forceinline__ void async_load16(unsigned int lds_off, const void* gaddr) {
    unsigned long long ga = (unsigned long long)(uintptr_t)gaddr;
    asm volatile("global_load_async_to_lds_b128 %0, %1, off"
                 :
                 : "v"(lds_off), "v"(ga)
                 : "memory");
}
__device__ __forceinline__ void wait_async0() {
    asm volatile("s_wait_asynccnt 0x0" ::: "memory");
}

// ---------------------------------------------------------------------------
// Kernel A: gather sampled pixel embeddings and L2-normalize over D.
// ---------------------------------------------------------------------------
__global__ void k_gather_norm(const float* __restrict__ pix,
                              const int* __restrict__ pidx,
                              float* __restrict__ sampled) {
    int id = blockIdx.x * blockDim.x + threadIdx.x;
    if (id >= NN * KK) return;
    int n = id / KK;
    int ii = pidx[id];
    const float* base = pix + (size_t)n * DD * (SS * SS) + ii;
    float v[DD];
    float ss = 0.0f;
#pragma unroll
    for (int d = 0; d < DD; ++d) {
        v[d] = base[(size_t)d * (SS * SS)];
        ss = fmaf(v[d], v[d], ss);
    }
    float sc = 1.0f / fmaxf(sqrtf(ss), 1e-6f);
    float* out = sampled + (size_t)id * DD;
#pragma unroll
    for (int d = 0; d < DD; ++d) out[d] = v[d] * sc;
}

// ---------------------------------------------------------------------------
// Kernel B: per (n,m,k): online max & sum-exp of z = 20*sim over all v.
// One block (512 threads) per (n,m); thread t owns row k=t; mesh staged in LDS.
// ---------------------------------------------------------------------------
__global__ void __launch_bounds__(512)
k_row_stats(const float* __restrict__ sampled, const float* __restrict__ mesh,
            float* __restrict__ rowMax, float* __restrict__ rowSum) {
    int nm = blockIdx.x;
    int n = nm / MM, mi = nm % MM;
    int k = threadIdx.x;
    __shared__ float lm[64 * DD];   // 4 KB
    float s[DD];
    const float* srow = sampled + ((size_t)n * KK + k) * DD;
#pragma unroll
    for (int d = 0; d < DD; ++d) s[d] = srow[d];
    const float* mbase = mesh + (size_t)mi * VV * DD;
    float mx = -3.4e38f, sm = 0.0f;
    for (int vb = 0; vb < VV; vb += 64) {
        int cnt = (VV - vb) < 64 ? (VV - vb) : 64;
        __syncthreads();
        for (int i = threadIdx.x; i < cnt * DD; i += blockDim.x)
            lm[i] = mbase[(size_t)vb * DD + i];
        __syncthreads();
        for (int j = 0; j < cnt; ++j) {
            float dot = 0.0f;
#pragma unroll
            for (int d = 0; d < DD; ++d) dot = fmaf(s[d], lm[j * DD + d], dot);
            float z = INV_T * dot;
            float nmx = fmaxf(mx, z);
            sm = sm * __expf(mx - nmx) + __expf(z - nmx);
            mx = nmx;
        }
    }
    rowMax[(size_t)nm * KK + k] = mx;
    rowSum[(size_t)nm * KK + k] = sm;
}

// ---------------------------------------------------------------------------
// Kernel C: per (n,m,v): online max & sum-exp of z over all k.
// ---------------------------------------------------------------------------
__global__ void __launch_bounds__(256)
k_col_stats(const float* __restrict__ sampled, const float* __restrict__ mesh,
            float* __restrict__ colMax, float* __restrict__ colSum) {
    int nm = blockIdx.x;
    int n = nm / MM, mi = nm % MM;
    __shared__ float ls[KK * DD];   // 32 KB
    for (int i = threadIdx.x; i < KK * DD; i += blockDim.x)
        ls[i] = sampled[(size_t)n * KK * DD + i];
    __syncthreads();
    int v = blockIdx.y * blockDim.x + threadIdx.x;
    if (v >= VP) return;
    if (v >= VV) {  // pad region: deterministic filler (never used for P/Q)
        colMax[(size_t)nm * VP + v] = 0.0f;
        colSum[(size_t)nm * VP + v] = 1.0f;
        return;
    }
    float mrow[DD];
    const float* mr = mesh + ((size_t)mi * VV + v) * DD;
#pragma unroll
    for (int d = 0; d < DD; ++d) mrow[d] = mr[d];
    float mx = -3.4e38f, sm = 0.0f;
    for (int k = 0; k < KK; ++k) {
        float dot = 0.0f;
#pragma unroll
        for (int d = 0; d < DD; ++d) dot = fmaf(mrow[d], ls[k * DD + d], dot);
        float z = INV_T * dot;
        float nmx = fmaxf(mx, z);
        sm = sm * __expf(mx - nmx) + __expf(z - nmx);
        mx = nmx;
    }
    colMax[(size_t)nm * VP + v] = mx;
    colSum[(size_t)nm * VP + v] = sm;
}

// ---------------------------------------------------------------------------
// Kernel D: materialize P (softmax over v) and Q (softmax over k) in f16,
// row-major [K, VP], zero-padded for v >= V.
// ---------------------------------------------------------------------------
__global__ void __launch_bounds__(256)
k_build_pq(const float* __restrict__ sampled, const float* __restrict__ mesh,
           const float* __restrict__ rowMax, const float* __restrict__ rowSum,
           const float* __restrict__ colMax, const float* __restrict__ colSum,
           _Float16* __restrict__ P, _Float16* __restrict__ Q,
           int n, int mi, int nm) {
    size_t id = (size_t)blockIdx.x * blockDim.x + threadIdx.x;
    if (id >= (size_t)KK * VP) return;
    int k = (int)(id / VP);
    int v = (int)(id % VP);
    if (v >= VV) { P[id] = (_Float16)0.0f; Q[id] = (_Float16)0.0f; return; }
    const float* srow = sampled + ((size_t)n * KK + k) * DD;
    const float* mr   = mesh + ((size_t)mi * VV + v) * DD;
    float dot = 0.0f;
#pragma unroll
    for (int d = 0; d < DD; ++d) dot = fmaf(srow[d], mr[d], dot);
    float z = INV_T * dot;
    float p = __expf(z - rowMax[(size_t)nm * KK + k]) * (1.0f / rowSum[(size_t)nm * KK + k]);
    float q = __expf(z - colMax[(size_t)nm * VP + v]) * (1.0f / colSum[(size_t)nm * VP + v]);
    P[id] = (_Float16)p;
    Q[id] = (_Float16)q;
}

// ---------------------------------------------------------------------------
// Kernel E: LDS-tiled WMMA GEMM  C[k,q] = sum_v P[k,v]*Q[q,v], fused with the
// distance weighting + squared-sum reduction.
//
// Block = 512 threads (16 waves) -> 64x64 output tile (4x4 grid of 16x16 wave
// tiles). Per step, 64 v-columns of P (A panel, 64x64h) and Q (B panel) are
// staged into LDS with GLOBAL_LOAD_ASYNC_TO_LDS_B128 (exactly 2 per thread),
// double buffered; s_wait_asynccnt + barrier, then each wave runs two
// v_wmma_f32_16x16x32_f16 from ds_load_b128 reads. The next panel's async DMA
// overlaps the current panel's WMMAs.
//
// A layout (16-bit 16x32 MxK, wave32): lane L -> row M=L%16; v16h elems 0..7 =
//   K=(L<16?0:8)+0..7 ; elems 8..15 = 16+(L<16?0:8)+0..7.
// B layout (32x16 KxN): lane L -> col N=L%16; elems 0..7 = K=(L<16?0:8)+0..7 ;
//   elems 8..15 = 16+(L<16?0:8)+0..7.
// C/D layout: VGPR r, lane L -> M = r + (L<16?0:8), N = L%16.
// ---------------------------------------------------------------------------
__global__ void __launch_bounds__(512)
k_cycle_wmma(const _Float16* __restrict__ P, const _Float16* __restrict__ Q,
             const int* __restrict__ pidx, float* __restrict__ partials,
             int n, int nm) {
    extern __shared__ _Float16 lds[];
    // Low 32 bits of the generic shared pointer == LDS byte offset.
    unsigned int smem_base = (unsigned int)(uintptr_t)lds;

    int t    = threadIdx.x;
    int lane = t & 31;
    int w    = t >> 5;            // wave 0..15
    int l16  = lane & 15;
    int hi   = (lane >> 4) & 1;
    int wi   = w >> 2;            // ktile within block 0..3
    int wj   = w & 3;             // qtile within block 0..3
    int kbase = blockIdx.x * 64;
    int qbase = blockIdx.y * 64;

    // Async staging assignment: thread t copies one 16B segment of A and one
    // of B per step. row = t/8 (0..63), seg = t%8 (8 x 16B = 64 halves).
    int srow = t >> 3;
    int seg  = t & 7;
    const _Float16* gArow = P + (size_t)(kbase + srow) * VP + seg * 8;
    const _Float16* gBrow = Q + (size_t)(qbase + srow) * VP + seg * 8;
    unsigned int ldsA = smem_base + (unsigned int)(srow * (ROWH * 2) + seg * 16);
    unsigned int ldsB = ldsA + PANEL_B;

    // LDS read bases (in halves) for this wave's 16x16 tile.
    int aRow = (16 * wi + l16) * ROWH;
    int bRow = PANEL_H + (16 * wj + l16) * ROWH;
    int aoff = hi ? 8 : 0;
    int boff = hi ? 8 : 0;

    const int steps = VP / 64;    // 123

    // Prologue: stage step 0 into buffer 0.
    async_load16(ldsA, gArow);
    async_load16(ldsB, gBrow);

    v8f c = {};
    for (int s = 0; s < steps; ++s) {
        wait_async0();            // this wave's panel writes for step s landed
        __syncthreads();          // all waves' writes landed; prev reads done
        if (s + 1 < steps) {      // stage step s+1 into the other buffer
            unsigned int bo = (unsigned int)(((s + 1) & 1) * BUF_B);
            const _Float16* ga = gArow + (size_t)(s + 1) * 64;
            const _Float16* gb = gBrow + (size_t)(s + 1) * 64;
            async_load16(ldsA + bo, ga);
            async_load16(ldsB + bo, gb);
        }
        const _Float16* buf = lds + (size_t)((s & 1) * (BUF_B / 2));
#pragma unroll
        for (int cc = 0; cc < 2; ++cc) {
            const _Float16* ar = buf + aRow + cc * 32;
            const _Float16* br = buf + bRow + cc * 32;
            hv8 a0 = *(const hv8*)(ar + aoff);
            hv8 a1 = *(const hv8*)(ar + 16 + aoff);
            hv8 b0 = *(const hv8*)(br + boff);
            hv8 b1 = *(const hv8*)(br + 16 + boff);
            v16h a = __builtin_shufflevector(a0, a1, 0,1,2,3,4,5,6,7,8,9,10,11,12,13,14,15);
            v16h b = __builtin_shufflevector(b0, b1, 0,1,2,3,4,5,6,7,8,9,10,11,12,13,14,15);
            c = __builtin_amdgcn_wmma_f32_16x16x32_f16(false, a, false, b,
                                                       (short)0, c, false, false);
        }
    }

    // Epilogue: weight each c element by squared pixel distance, square, sum.
    int q  = qbase + 16 * wj + l16;
    int qi = pidx[n * KK + q];
    float qr = (float)(qi / SS), qc = (float)(qi % SS);
    float acc = 0.0f;
#pragma unroll
    for (int r = 0; r < 8; ++r) {
        int kk = kbase + 16 * wi + r + (hi ? 8 : 0);
        int ki = pidx[n * KK + kk];
        float dr = (float)(ki / SS) - qr;
        float dc = (float)(ki % SS) - qc;
        float dist = fmaf(dr, dr, dc * dc);
        float wv = dist * c[r];
        acc = fmaf(wv, wv, acc);
    }
    for (int off = 16; off > 0; off >>= 1) acc += __shfl_xor(acc, off);
    if (lane == 0)
        partials[(size_t)nm * 1024 + ((blockIdx.x * 8 + blockIdx.y) * 16 + w)] = acc;
}

// ---------------------------------------------------------------------------
// Kernel F: loss = mean over 32 (n,m) of sqrt(sum of 1024 partials).
// ---------------------------------------------------------------------------
__global__ void __launch_bounds__(256)
k_finalize(const float* __restrict__ partials, float* __restrict__ out) {
    __shared__ float red[256];
    int t = threadIdx.x;
    float tot = 0.0f;
    for (int nm = 0; nm < NN * MM; ++nm) {
        float s = 0.0f;
        for (int i = t; i < 1024; i += 256) s += partials[(size_t)nm * 1024 + i];
        red[t] = s;
        __syncthreads();
        for (int o = 128; o > 0; o >>= 1) {
            if (t < o) red[t] += red[t + o];
            __syncthreads();
        }
        if (t == 0) tot += sqrtf(red[0]);
        __syncthreads();
    }
    if (t == 0) out[0] = tot / (float)(NN * MM);
}

// ---------------------------------------------------------------------------
// Host launcher
// ---------------------------------------------------------------------------
static inline size_t align_up(size_t x, size_t a) { return (x + a - 1) & ~(a - 1); }

extern "C" void kernel_launch(void* const* d_in, const int* in_sizes, int n_in,
                              void* d_out, int out_size, void* d_ws, size_t ws_size,
                              hipStream_t stream) {
    const float* pix  = (const float*)d_in[0];   // [N,D,S,S]
    const float* mesh = (const float*)d_in[1];   // [M,V,D]
    const int*   pidx = (const int*)d_in[2];     // [N,K]
    float* out = (float*)d_out;

    char* ws = (char*)d_ws;
    size_t off = 0;
    float* sampled = (float*)(ws + off); off = align_up(off + (size_t)NN * KK * DD * 4, 256);
    float* rowMax  = (float*)(ws + off); off = align_up(off + (size_t)NN * MM * KK * 4, 256);
    float* rowSum  = (float*)(ws + off); off = align_up(off + (size_t)NN * MM * KK * 4, 256);
    float* colMax  = (float*)(ws + off); off = align_up(off + (size_t)NN * MM * VP * 4, 256);
    float* colSum  = (float*)(ws + off); off = align_up(off + (size_t)NN * MM * VP * 4, 256);
    float* partials= (float*)(ws + off); off = align_up(off + (size_t)NN * MM * 1024 * 4, 256);
    _Float16* P    = (_Float16*)(ws + off); off = align_up(off + (size_t)KK * VP * 2, 256);
    _Float16* Q    = (_Float16*)(ws + off); off = align_up(off + (size_t)KK * VP * 2, 256);
    (void)ws_size; (void)in_sizes; (void)n_in; (void)out_size;

    // A: gather + normalize
    k_gather_norm<<<(NN * KK + 255) / 256, 256, 0, stream>>>(pix, pidx, sampled);
    // B: row (over-v) softmax stats for all (n,m)
    k_row_stats<<<NN * MM, 512, 0, stream>>>(sampled, mesh, rowMax, rowSum);
    // C: col (over-k) softmax stats for all (n,m)
    {
        dim3 g(NN * MM, (VP + 255) / 256);
        k_col_stats<<<g, 256, 0, stream>>>(sampled, mesh, colMax, colSum);
    }
    // Per (n,m): build f16 P/Q, then async-LDS-tiled WMMA GEMM + fused loss.
    const int pq_blocks = (int)(((size_t)KK * VP + 255) / 256);
    for (int nm = 0; nm < NN * MM; ++nm) {
        int n = nm / MM, mi = nm % MM;
        k_build_pq<<<pq_blocks, 256, 0, stream>>>(sampled, mesh, rowMax, rowSum,
                                                  colMax, colSum, P, Q, n, mi, nm);
        dim3 g(KK / 64, KK / 64);   // (8, 8) blocks of 512 threads
        k_cycle_wmma<<<g, 512, LDS_TOTAL, stream>>>(P, Q, pidx, partials, n, nm);
    }
    // F: sqrt + mean
    k_finalize<<<1, 256, 0, stream>>>(partials, out);
}
